// EAP_v1_38972533243922
// MI455X (gfx1250) — matrix operand order repaired and checked by reading, not compiled, not run on hardware
//
#include <hip/hip_runtime.h>
#include <hip/hip_bf16.h>

typedef __attribute__((ext_vector_type(16))) _Float16 v16h;
typedef __attribute__((ext_vector_type(8)))  float    v8f;

#define HW    16384   // 128*128 pixels per batch
#define NPIX  4096    // 64*64 downsampled positions per batch
#define CIN   256
#define C3    768     // 3*256 qkv channels
#define OD    256
#define NB    4
#define EPSLN 1e-6f

// ---------------------------------------------------------------------------
// Kernel 1: qkv 1x1 conv as WMMA GEMM.
//   qkv_t[b][p][o] = sum_c x[b][c][p] * w_qkv[o][c]
// M = pixels, N = out-channels, K = input channels.
// One wave computes a 16(M) x 64(N) tile (4 accumulators) so the expensive
// A fragment (coalesced scalar loads of x) is reused by 4 WMMAs.
// A-frag (16x32 f16): lane l: m = l&15 (pixel), g = l>>4,
//   elem i -> k = 16g+i
// B-frag (32x16 f16): lane l: n = l&15 (out ch), elem i -> k = 16g+i
//   (16 contiguous floats of a weight row -> 4x float4)
// D (16x16 f32): lane l: n = l&15, reg r -> m = r + 8g.
// ---------------------------------------------------------------------------
__global__ void qkv_gemm_wmma(const float* __restrict__ x,      // [B][256][HW]
                              const float* __restrict__ w_qkv,  // [768][256]
                              float* __restrict__ qkv_t) {      // [B][HW][768]
  const int wave  = threadIdx.x >> 5;
  const int l     = threadIdx.x & 31;
  const int g     = l >> 4;
  const int mn    = l & 15;
  const int oBase = (blockIdx.x * 4 + wave) * 64;   // 64-wide out-channel tile
  const int pBase = blockIdx.y * 16;                // pixel tile
  const int b     = blockIdx.z;

  const float* xb = x + (size_t)b * CIN * HW;       // [c][p]
  const int    p  = pBase + mn;                     // A lane row (pixel)

  const float* wrow[4];
#pragma unroll
  for (int t = 0; t < 4; ++t) {
    wrow[t] = w_qkv + (size_t)(oBase + 16 * t + mn) * CIN;
    __builtin_prefetch(wrow[t], 0, 3);              // global_prefetch_b8
  }

  v8f acc[4] = {};
  for (int k0 = 0; k0 < CIN; k0 += 32) {
    v16h a;
#pragma unroll
    for (int i = 0; i < 16; ++i) {
      const int k = k0 + 16 * g + i;
      a[i] = (_Float16)xb[(size_t)k * HW + p];
    }
#pragma unroll
    for (int t = 0; t < 4; ++t) {
      const float4* wp = (const float4*)(wrow[t] + k0 + 16 * g);
      float wv[16];
      *(float4*)(wv + 0)  = wp[0];
      *(float4*)(wv + 4)  = wp[1];
      *(float4*)(wv + 8)  = wp[2];
      *(float4*)(wv + 12) = wp[3];
      v16h bf;
#pragma unroll
      for (int i = 0; i < 16; ++i) bf[i] = (_Float16)wv[i];
      acc[t] = __builtin_amdgcn_wmma_f32_16x16x32_f16(false, a, false, bf,
                                                      (short)0, acc[t],
                                                      false, false);
    }
  }

  float* outb = qkv_t + (size_t)b * HW * C3;
#pragma unroll
  for (int r = 0; r < 8; ++r) {
    const size_t rowOff = (size_t)(pBase + r + 8 * g) * C3;
#pragma unroll
    for (int t = 0; t < 4; ++t)
      outb[rowOff + oBase + 16 * t + mn] = acc[t][r];  // 64B coalesced / group
  }
}

// ---------------------------------------------------------------------------
// Kernel 2: fused dwconv(q)+LN(q) / per-pixel LN(k,v gather) / 9-way softmax
// One block = one (b, n) position, 256 threads = 256 channels, wave == head.
// Unfold+channels-first LN == per-source-pixel LN over C; OOB windows give the
// all-zero vector whose LN degenerates to the bias (0) -- handled naturally.
// ---------------------------------------------------------------------------
__global__ void attn_fused(const float* __restrict__ qkv_t,   // [B][HW][768]
                           const float* __restrict__ w_dwq,   // [256][1][3][3]
                           const float* __restrict__ lnqw, const float* __restrict__ lnqb,
                           const float* __restrict__ lnkw, const float* __restrict__ lnkb,
                           const float* __restrict__ lnvw, const float* __restrict__ lnvb,
                           float* __restrict__ attn_out) {    // [B][NPIX][256]
  const int c = threadIdx.x;            // channel
  const int n = blockIdx.x;             // downsampled position
  const int b = blockIdx.y;
  const int y = n >> 6, xc = n & 63;
  const float* base = qkv_t + (size_t)b * HW * C3;

  float qv[9], kv[9], vv[9];
#pragma unroll
  for (int j = 0; j < 9; ++j) {
    const int i = j / 3, jj = j % 3;
    const int sy = 2 * y + i - 1, sx = 2 * xc + jj - 1;
    const bool valid = ((unsigned)sy < 128u) && ((unsigned)sx < 128u);
    if (valid) {
      const float* pp = base + (size_t)(sy * 128 + sx) * C3;
      qv[j] = pp[c];                    // coalesced across the 256 threads
      kv[j] = pp[256 + c];
      vv[j] = pp[512 + c];
    } else {
      qv[j] = 0.f; kv[j] = 0.f; vv[j] = 0.f;
    }
  }

  // depthwise 3x3 stride-2 conv for q
  float qd = 0.f;
#pragma unroll
  for (int j = 0; j < 9; ++j) qd += w_dwq[c * 9 + j] * qv[j];

  // 19 channel reductions (sum, sumsq over 256 channels):
  // [0]=qd, [1..9]=k window pixels, [10..18]=v window pixels
  float sv[19], sq[19];
  sv[0] = qd;
#pragma unroll
  for (int j = 0; j < 9; ++j) { sv[1 + j] = kv[j]; sv[10 + j] = vv[j]; }
#pragma unroll
  for (int i = 0; i < 19; ++i) sq[i] = sv[i] * sv[i];

#pragma unroll
  for (int i = 0; i < 19; ++i) {
#pragma unroll
    for (int off = 16; off > 0; off >>= 1) {
      sv[i] += __shfl_xor(sv[i], off, 32);
      sq[i] += __shfl_xor(sq[i], off, 32);
    }
  }
  // transposed partial layout: [quantity][wave] -> float4-friendly finalize
  __shared__ float redS[19][8];
  __shared__ float redQ[19][8];
  __shared__ float muS[19], rsS[19];
  const int lane = threadIdx.x & 31, wid = threadIdx.x >> 5;
  if (lane == 0) {
#pragma unroll
    for (int i = 0; i < 19; ++i) { redS[i][wid] = sv[i]; redQ[i][wid] = sq[i]; }
  }
  __syncthreads();
  if (threadIdx.x < 19) {
    const int i = threadIdx.x;
    float4 s0 = *(const float4*)&redS[i][0];
    float4 s1 = *(const float4*)&redS[i][4];
    float4 q0 = *(const float4*)&redQ[i][0];
    float4 q1 = *(const float4*)&redQ[i][4];
    const float s  = (s0.x + s0.y) + (s0.z + s0.w) + (s1.x + s1.y) + (s1.z + s1.w);
    const float ss = (q0.x + q0.y) + (q0.z + q0.w) + (q1.x + q1.y) + (q1.z + q1.w);
    const float m   = s * (1.f / 256.f);
    const float var = ss * (1.f / 256.f) - m * m;   // biased variance
    muS[i] = m;
    rsS[i] = rsqrtf(var + EPSLN);
  }
  __syncthreads();

  const float qdn = lnqw[c] * (qd - muS[0]) * rsS[0] + lnqb[c];
  float kn[9], vn[9];
#pragma unroll
  for (int j = 0; j < 9; ++j) {
    kn[j] = lnkw[c] * (kv[j] - muS[1 + j])  * rsS[1 + j]  + lnkb[c];
    vn[j] = lnvw[c] * (vv[j] - muS[10 + j]) * rsS[10 + j] + lnvb[c];
  }

  // per-head attention: this wave's 32 lanes are exactly one head's channels
  const float scale = 0.1767766952966369f;  // 32^-0.5
  float lg[9];
#pragma unroll
  for (int j = 0; j < 9; ++j) {
    float d = qdn * kn[j];
#pragma unroll
    for (int off = 16; off > 0; off >>= 1) d += __shfl_xor(d, off, 32);
    lg[j] = d * scale;
  }
  float mx = lg[0];
#pragma unroll
  for (int j = 1; j < 9; ++j) mx = fmaxf(mx, lg[j]);
  float e[9], es = 0.f;
#pragma unroll
  for (int j = 0; j < 9; ++j) { e[j] = expf(lg[j] - mx); es += e[j]; }
  const float inv = 1.f / es;
  float oc = 0.f;
#pragma unroll
  for (int j = 0; j < 9; ++j) oc += e[j] * inv * vn[j];

  attn_out[((size_t)(b * NPIX + n)) * OD + c] = oc;
}

// ---------------------------------------------------------------------------
// Kernel 3: output projection as WMMA GEMM (+bias), rows = B*NPIX = 16384.
//   out[r][o] = sum_c attn_out[r][c] * w_proj[o][c] + b_proj[o]
// One wave computes 16 rows x 64 out-channels (4 accumulators, shared A).
// ---------------------------------------------------------------------------
__global__ void proj_gemm_wmma(const float* __restrict__ attn_out, // [16384][256]
                               const float* __restrict__ w_proj,   // [256][256]
                               const float* __restrict__ b_proj,   // [256]
                               float* __restrict__ out) {          // [16384][256]+2
  const int wave  = threadIdx.x >> 5;
  const int l     = threadIdx.x & 31;
  const int g     = l >> 4;
  const int mn    = l & 15;
  const int oBase = wave * 64;                 // 4 waves cover all 256 channels
  const int rBase = blockIdx.x * 16;

  const float* arow = attn_out + (size_t)(rBase + mn) * OD;
  const float* wrow[4];
#pragma unroll
  for (int t = 0; t < 4; ++t) {
    wrow[t] = w_proj + (size_t)(oBase + 16 * t + mn) * OD;
    __builtin_prefetch(wrow[t], 0, 3);
  }

  v8f acc[4] = {};
  for (int k0 = 0; k0 < OD; k0 += 32) {
    // A elems: 16 contiguous floats at k0+16g
    float av[16];
    const float4* a0 = (const float4*)(arow + k0 + 16 * g);
    const float4* a1 = (const float4*)(arow + k0 + 16 * g + 8);
    *(float4*)(av + 0)  = a0[0];
    *(float4*)(av + 4)  = a0[1];
    *(float4*)(av + 8)  = a1[0];
    *(float4*)(av + 12) = a1[1];
    v16h a;
#pragma unroll
    for (int i = 0; i < 16; ++i) a[i] = (_Float16)av[i];

#pragma unroll
    for (int t = 0; t < 4; ++t) {
      const float4* wp = (const float4*)(wrow[t] + k0 + 16 * g);
      float wv[16];
      *(float4*)(wv + 0)  = wp[0];
      *(float4*)(wv + 4)  = wp[1];
      *(float4*)(wv + 8)  = wp[2];
      *(float4*)(wv + 12) = wp[3];
      v16h bf;
#pragma unroll
      for (int i = 0; i < 16; ++i) bf[i] = (_Float16)wv[i];
      acc[t] = __builtin_amdgcn_wmma_f32_16x16x32_f16(false, a, false, bf,
                                                      (short)0, acc[t],
                                                      false, false);
    }
  }

#pragma unroll
  for (int r = 0; r < 8; ++r) {
    const size_t rowOff = (size_t)(rBase + r + 8 * g) * OD;
#pragma unroll
    for (int t = 0; t < 4; ++t) {
      const int o = oBase + 16 * t + mn;
      out[rowOff + o] = acc[t][r] + b_proj[o];
    }
  }

  // tuple tail: reference returns (out, h, w) with h = w = 64
  if (blockIdx.x == 0 && threadIdx.x == 0) {
    out[(size_t)NB * NPIX * OD + 0] = 64.0f;
    out[(size_t)NB * NPIX * OD + 1] = 64.0f;
  }
}

// ---------------------------------------------------------------------------
extern "C" void kernel_launch(void* const* d_in, const int* in_sizes, int n_in,
                              void* d_out, int out_size, void* d_ws, size_t ws_size,
                              hipStream_t stream) {
  const float* x      = (const float*)d_in[0];
  const float* w_qkv  = (const float*)d_in[1];
  const float* w_dwq  = (const float*)d_in[2];
  const float* ln_q_w = (const float*)d_in[3];
  const float* ln_q_b = (const float*)d_in[4];
  const float* ln_k_w = (const float*)d_in[5];
  const float* ln_k_b = (const float*)d_in[6];
  const float* ln_v_w = (const float*)d_in[7];
  const float* ln_v_b = (const float*)d_in[8];
  const float* w_proj = (const float*)d_in[9];
  const float* b_proj = (const float*)d_in[10];
  float* out = (float*)d_out;

  // workspace: qkv_t [4][16384][768] f32 (192 MB) + attn [4][4096][256] (16 MB)
  float* qkv_t = (float*)d_ws;
  float* attn  = qkv_t + (size_t)NB * HW * C3;

  {
    dim3 grid(3, 1024, NB);    // 12 o-quads (4/block, 64 ch each) x 1024 p-tiles
    dim3 block(128);           // 4 waves
    qkv_gemm_wmma<<<grid, block, 0, stream>>>(x, w_qkv, qkv_t);
  }
  {
    dim3 grid(NPIX, NB);
    dim3 block(256);
    attn_fused<<<grid, block, 0, stream>>>(qkv_t, w_dwq,
                                           ln_q_w, ln_q_b, ln_k_w, ln_k_b,
                                           ln_v_w, ln_v_b, attn);
  }
  {
    dim3 grid(1024);           // 1024 row-tiles; 4 waves cover 256 channels
    dim3 block(128);
    proj_gemm_wmma<<<grid, block, 0, stream>>>(attn, w_proj, b_proj, out);
  }
}
